// Encoder_26036091748684
// MI455X (gfx1250) — matrix operand order repaired and checked by reading, not compiled, not run on hardware
//
#include <hip/hip_runtime.h>
#include <hip/hip_bf16.h>

typedef __attribute__((ext_vector_type(16))) _Float16 v16h;
typedef __attribute__((ext_vector_type(8)))  _Float16 v8h;
typedef __attribute__((ext_vector_type(8)))  float    v8f;
typedef __attribute__((ext_vector_type(4)))  float    v4f;

// ---------------- constants (concat layout) ----------------
// concat row offsets inside agg_w (1348 x 256):
//   [0,512)    species_tbl  -> fusedS
//   [512,640)  ability_tbl  -> fusedA
//   [640,896)  item_tbl     -> fusedI
//   [896,1080) 13 scalar one-hots (cumulative offsets below)
//   [1080,1171) 7 boost one-hots (13 each)
//   [1171,1315) 9 x 16 VOL bits
//   [1315,1347) 2 x 16 TC bits
//   [1347]     hp/31 scaled row
#define OFF_SCALAR 896
#define OFF_BOOST  1080
#define OFF_VOL    1171
#define OFF_TC     1315
#define ROW_HP     1347

__constant__ int kScalarCum[13] = {0,101,103,105,137,140,148,164,166,168,170,178,182};

// ---------------- kernel 1: fused categorical tables ----------------
// fusedS[s][j] = agg_b[j] + species_emb[s][j] + sum_k species_tbl[s][k]*agg_w[k][j]
// fusedA[a][j] = ability_emb[a][j] + sum_k ability_tbl[a][k]*agg_w[512+k][j]
// fusedI[i][j] = item_emb[i][j]    + sum_k item_tbl[i][k]*agg_w[640+k][j]
__global__ __launch_bounds__(256) void build_fused(
    const float* __restrict__ species_tbl, const float* __restrict__ ability_tbl,
    const float* __restrict__ item_tbl,
    const float* __restrict__ species_emb, const float* __restrict__ ability_emb,
    const float* __restrict__ item_emb,
    const float* __restrict__ agg_w, const float* __restrict__ agg_b,
    float* __restrict__ fusedS, float* __restrict__ fusedA, float* __restrict__ fusedI)
{
    int r = blockIdx.x;        // 0..895
    int j = threadIdx.x;       // 0..255
    const float* tbl; const float* emb; const float* wb; float* dst; int K; float acc;
    if (r < 512)      { tbl = species_tbl + r*512;       emb = species_emb + r*256;
                        wb = agg_w;           dst = fusedS + r*256;       K = 512; acc = agg_b[j]; }
    else if (r < 640) { int rr = r-512; tbl = ability_tbl + rr*128; emb = ability_emb + rr*256;
                        wb = agg_w + 512*256; dst = fusedA + rr*256;     K = 128; acc = 0.0f; }
    else              { int rr = r-640; tbl = item_tbl + rr*256;    emb = item_emb + rr*256;
                        wb = agg_w + 640*256; dst = fusedI + rr*256;     K = 256; acc = 0.0f; }
    acc += emb[j];
    for (int k = 0; k < K; ++k)
        acc = fmaf(tbl[k], wb[k*256 + j], acc);   // tbl[k] is wave-uniform, agg_w coalesced
    dst[j] = acc;
}

// ---------------- kernel 2: mlp_w -> f16, transposed (WhT[n][k]) ----------------
__global__ __launch_bounds__(256) void build_wt(const float* __restrict__ mlp_w,
                                                _Float16* __restrict__ WhT)
{
    int ncol = blockIdx.x;     // 0..255
    int k    = threadIdx.x;    // 0..255
    WhT[ncol*256 + k] = (_Float16)mlp_w[k*256 + ncol];
}

// ---------------- kernel 3: per-entity gather/accumulate, ReLU, f16 ----------------
// One wave per entity. Lane l accumulates output dims [l*8, l*8+8).
// Feature values are moved to SGPRs (readfirstlane) so bit tests become
// scalar branches and every row fetch is SGPR-base + lane*32B (coalesced 1KB).
__global__ __launch_bounds__(256) void encode_accum(
    const int*   __restrict__ entity,
    const float* __restrict__ agg_w,        // 1348 x 256
    const float* __restrict__ actions_emb,  // 512 x 256
    const float* __restrict__ fusedS,
    const float* __restrict__ fusedA,
    const float* __restrict__ fusedI,
    _Float16*    __restrict__ Xr,           // n x 256 (f16, post-ReLU)
    int n)
{
    int wave = threadIdx.x >> 5;
    int lane = threadIdx.x & 31;
    int e = blockIdx.x * 8 + wave;
    if (e >= n) return;
    const int* ent = entity + (long)e * 38;

    v4f a0, a1;
    int sp = __builtin_amdgcn_readfirstlane(ent[0]);
    int ab = __builtin_amdgcn_readfirstlane(ent[1]);
    int it = __builtin_amdgcn_readfirstlane(ent[2]);
    {
        const v4f* p = (const v4f*)(fusedS + sp*256);
        a0 = p[lane*2]; a1 = p[lane*2 + 1];
    }
    {
        const v4f* p = (const v4f*)(fusedA + ab*256);
        a0 += p[lane*2]; a1 += p[lane*2 + 1];
    }
    {
        const v4f* p = (const v4f*)(fusedI + it*256);
        a0 += p[lane*2]; a1 += p[lane*2 + 1];
    }
    // 13 scalar one-hots
    #pragma unroll
    for (int i = 0; i < 13; ++i) {
        int v = __builtin_amdgcn_readfirstlane(ent[3 + i]);
        const v4f* p = (const v4f*)(agg_w + (long)(OFF_SCALAR + kScalarCum[i] + v) * 256);
        a0 += p[lane*2]; a1 += p[lane*2 + 1];
    }
    // 7 boost one-hots (13 classes each)
    #pragma unroll
    for (int i = 0; i < 7; ++i) {
        int v = __builtin_amdgcn_readfirstlane(ent[16 + i]);
        const v4f* p = (const v4f*)(agg_w + (long)(OFF_BOOST + 13*i + v) * 256);
        a0 += p[lane*2]; a1 += p[lane*2 + 1];
    }
    // 9 x 16 VOL bit rows (scalar while-loop over set bits)
    #pragma unroll
    for (int w = 0; w < 9; ++w) {
        int v = __builtin_amdgcn_readfirstlane(ent[23 + w]) & 0xFFFF;
        while (v) {
            int b = __builtin_ctz(v); v &= v - 1;
            const v4f* p = (const v4f*)(agg_w + (long)(OFF_VOL + 16*w + b) * 256);
            a0 += p[lane*2]; a1 += p[lane*2 + 1];
        }
    }
    // 2 x 16 TC bit rows
    #pragma unroll
    for (int w = 0; w < 2; ++w) {
        int v = __builtin_amdgcn_readfirstlane(ent[32 + w]) & 0xFFFF;
        while (v) {
            int b = __builtin_ctz(v); v &= v - 1;
            const v4f* p = (const v4f*)(agg_w + (long)(OFF_TC + 16*w + b) * 256);
            a0 += p[lane*2]; a1 += p[lane*2 + 1];
        }
    }
    // hp/31 scaled row
    {
        int hp = __builtin_amdgcn_readfirstlane(ent[6]);
        float s = (float)hp * (1.0f / 31.0f);
        const v4f* p = (const v4f*)(agg_w + (long)ROW_HP * 256);
        a0 += s * p[lane*2]; a1 += s * p[lane*2 + 1];
    }
    // 4 action embedding rows
    #pragma unroll
    for (int i = 0; i < 4; ++i) {
        int m = __builtin_amdgcn_readfirstlane(ent[34 + i]);
        const v4f* p = (const v4f*)(actions_emb + (long)m * 256);
        a0 += p[lane*2]; a1 += p[lane*2 + 1];
    }
    // ReLU -> f16, one 16B store per lane
    v8h o;
    #pragma unroll
    for (int i = 0; i < 4; ++i) {
        o[i]     = (_Float16)fmaxf(a0[i], 0.0f);
        o[4 + i] = (_Float16)fmaxf(a1[i], 0.0f);
    }
    *(v8h*)(Xr + (long)e * 256 + lane * 8) = o;
}

// ---------------- kernel 4: Y = ReLU(X) @ mlp_w + b, masked  (WMMA) ----------------
// Register-blocked: one wave computes 1 M-tile x 4 N-tiles (16x64 of output).
// Per K-step: 2 A loads feed 4 WMMAs (A reused across the 4 N-tiles),
// 8 B loads come from the 128KB L2-resident WhT.
// Block = 4 waves covering full N=256 for one M-tile; grid = n/16.
// A fragment (16x32 f16): lane l -> row m=l&15, halves k = kb + (l>>4)*8 (+16).
// B fragment (32x16 f16): lane l -> col n=l&15, halves k = kb + (l>>4)*8 (+16)
//                         (WhT is stored N-major so this is 2 contiguous 16B loads).
__global__ __launch_bounds__(128) void mlp_wmma(
    const _Float16* __restrict__ Xr,     // n x 256
    const _Float16* __restrict__ WhT,    // 256(n) x 256(k)
    const float*    __restrict__ mlp_b,
    const int*      __restrict__ entity,
    float*          __restrict__ out,    // n x 256
    int n)
{
    int wave = threadIdx.x >> 5;     // N-tile group 0..3 (tiles 4w..4w+3)
    int lane = threadIdx.x & 31;
    int tileM = blockIdx.x;
    int ln = lane & 15;
    int h  = lane >> 4;

    int rowA = tileM * 16 + ln;
    if (rowA >= n) rowA = n - 1;                    // select, no EXEC divergence
    const _Float16* aptr = Xr + (long)rowA * 256;
    const _Float16* bptr[4];
    #pragma unroll
    for (int t = 0; t < 4; ++t)
        bptr[t] = WhT + (long)((wave * 4 + t) * 16 + ln) * 256;

    v8f acc[4] = {};
    #pragma unroll
    for (int kb = 0; kb < 256; kb += 32) {
        v8h alo = *(const v8h*)(aptr + kb + h * 8);
        v8h ahi = *(const v8h*)(aptr + kb + 16 + h * 8);
        v16h A = __builtin_shufflevector(alo, ahi, 0,1,2,3,4,5,6,7,8,9,10,11,12,13,14,15);
        #pragma unroll
        for (int t = 0; t < 4; ++t) {
            v8h blo = *(const v8h*)(bptr[t] + kb + h * 8);
            v8h bhi = *(const v8h*)(bptr[t] + kb + 16 + h * 8);
            v16h B = __builtin_shufflevector(blo, bhi, 0,1,2,3,4,5,6,7,8,9,10,11,12,13,14,15);
            acc[t] = __builtin_amdgcn_wmma_f32_16x16x32_f16(
                         /*neg_a=*/false, A, /*neg_b=*/false, B,
                         /*c_mod=*/(short)0, acc[t], /*reuse_a=*/false, /*reuse_b=*/false);
        }
    }

    // species mask per output row (shared across the 4 N-tiles)
    int spv[8];
    #pragma unroll
    for (int i = 0; i < 8; ++i) {
        int m = tileM * 16 + h * 8 + i;
        spv[i] = (m < n) ? entity[(long)m * 38] : 0;   // m>=n rows never stored
    }

    #pragma unroll
    for (int t = 0; t < 4; ++t) {
        int ncol = (wave * 4 + t) * 16 + ln;
        float bias = mlp_b[ncol];
        #pragma unroll
        for (int i = 0; i < 8; ++i) {
            int m = tileM * 16 + h * 8 + i;
            if (m < n)
                out[(long)m * 256 + ncol] = (spv[i] < 2) ? 0.0f : (acc[t][i] + bias);
        }
    }
}

// ---------------- launch ----------------
extern "C" void kernel_launch(void* const* d_in, const int* in_sizes, int n_in,
                              void* d_out, int out_size, void* d_ws, size_t ws_size,
                              hipStream_t stream) {
    const int*   entity      = (const int*)  d_in[0];
    const float* species_tbl = (const float*)d_in[1];
    const float* ability_tbl = (const float*)d_in[2];
    const float* item_tbl    = (const float*)d_in[3];
    const float* species_emb = (const float*)d_in[4];
    const float* ability_emb = (const float*)d_in[5];
    const float* item_emb    = (const float*)d_in[6];
    const float* actions_emb = (const float*)d_in[7];
    const float* agg_w       = (const float*)d_in[8];
    const float* agg_b       = (const float*)d_in[9];
    const float* mlp_w       = (const float*)d_in[10];
    const float* mlp_b       = (const float*)d_in[11];
    float* out = (float*)d_out;

    const int n = in_sizes[0] / 38;   // number of entities (65536)

    // workspace layout (all 16B-aligned); total ~34.6 MB
    char* ws = (char*)d_ws;
    float*    fusedS = (float*)   (ws);                 // 512*256*4  = 512 KB
    float*    fusedA = (float*)   (ws + 524288);        // 128*256*4  = 128 KB
    float*    fusedI = (float*)   (ws + 655360);        // 256*256*4  = 256 KB
    _Float16* WhT    = (_Float16*)(ws + 917504);        // 256*256*2  = 128 KB
    _Float16* Xr     = (_Float16*)(ws + 1048576);       // n*256*2    = 32 MB

    build_fused<<<896, 256, 0, stream>>>(species_tbl, ability_tbl, item_tbl,
                                         species_emb, ability_emb, item_emb,
                                         agg_w, agg_b, fusedS, fusedA, fusedI);
    build_wt<<<256, 256, 0, stream>>>(mlp_w, WhT);
    encode_accum<<<(n + 7) / 8, 256, 0, stream>>>(entity, agg_w, actions_emb,
                                                  fusedS, fusedA, fusedI, Xr, n);
    mlp_wmma<<<(n + 15) / 16, 128, 0, stream>>>(Xr, WhT, mlp_b, entity, out, n);
}
